// Model_44598940401760
// MI455X (gfx1250) — matrix operand order repaired and checked by reading, not compiled, not run on hardware
//
#include <hip/hip_runtime.h>
#include <hip/hip_bf16.h>

typedef _Float16 f16;
typedef __attribute__((ext_vector_type(16))) _Float16 v16h;
typedef __attribute__((ext_vector_type(8)))  _Float16 v8h;
typedef __attribute__((ext_vector_type(8)))  float    v8f;

#define ROWS_PER_WG 64
#define THREADS     256
#define NWAVES      8

// ---- LDS layout (bytes). Pitches in f16 elements, padded +8 for bank spread,
// ---- all multiples of 8 f16 => every row base is 16-byte aligned.
#define XBUF_PITCH 968   // 960 (=ceil(937/32)*32) + 8
#define H512_PITCH 520
#define H256_PITCH 264
#define H128_PITCH 136
#define H64_PITCH  72

#define XBUF_BYTES (ROWS_PER_WG * XBUF_PITCH * 2)   // 123904
#define HBUF_BYTES (ROWS_PER_WG * H512_PITCH * 2)   // 66560
#define HA_OFF     XBUF_BYTES
#define HB_OFF     (HA_OFF + HBUF_BYTES)
#define LDS_TOTAL  (HB_OFF + HBUF_BYTES)            // 257024 < 320KB WGP LDS

// Head scratch aliases the xbuf region (x dead after layer 1).
#define V1_OFF 0
#define A1_OFF (ROWS_PER_WG * H128_PITCH * 2)                 // 17408
#define V2_OFF (A1_OFF + ROWS_PER_WG * H128_PITCH * 2)        // 34816
#define A2_OFF (V2_OFF + ROWS_PER_WG * H64_PITCH * 2)         // 44032
#define VF_OFF (A2_OFF + ROWS_PER_WG * H64_PITCH * 2)         // 53248
#define AF_OFF (VF_OFF + ROWS_PER_WG * 16 * 4)                // 57344
static_assert(AF_OFF + ROWS_PER_WG * 16 * 4 <= XBUF_BYTES, "head scratch fits in xbuf");

struct Params {
    const float* x;
    const unsigned char* suicides;   // jnp bool -> 1 byte/elem
    const int* invalid;
    float* out;
    const f16* w[13];
    const float* b[13];
};

// ---------------------------------------------------------------------------
// Pack f32 weight [K,N] row-major -> f16 Wp[kb][n][kk] with K padded to 32,
// N padded to 16 (zeros in padding). B-fragment for tile (k0,n0), lane L is
// then 16 contiguous f16 at Wp + (k0/32)*Np*32 + (n0+(L&15))*32 + (L&16?16:0).
// ---------------------------------------------------------------------------
__global__ void pack_weights(const float* __restrict__ W, f16* __restrict__ Wp,
                             int K, int N, int Kp, int Np) {
    int total = Kp * Np;
    for (int i = blockIdx.x * blockDim.x + threadIdx.x; i < total;
         i += gridDim.x * blockDim.x) {
        int kb  = i / (Np * 32);
        int rem = i - kb * (Np * 32);
        int n   = rem >> 5;
        int kk  = rem & 31;
        int k   = kb * 32 + kk;
        float v = (k < K && n < N) ? W[k * N + n] : 0.0f;
        Wp[i] = (f16)v;
    }
}

// ---------------------------------------------------------------------------
// One dense layer: out[64 x Np] = act(A[64 x 32*KB] @ W + b)
// A in LDS (f16, pitch aPitch), W packed in global, out to LDS (f16 or f32).
// Wave-tiled: 16x16 output tiles, K-loop of v_wmma_f32_16x16x32_f16.
// ---------------------------------------------------------------------------
template<int KB, bool RELU, bool OUTF32>
__device__ __forceinline__ void layer_gemm(
    const f16* A, int aPitch,
    const f16* __restrict__ Wp, const float* __restrict__ bias,
    int Nvalid, int Np,
    f16* outH, float* outF, int outPitch,
    int lane, int wave)
{
    const int mTiles = ROWS_PER_WG / 16;          // 4
    const int T = mTiles * (Np >> 4);
    for (int t = wave; t < T; t += NWAVES) {      // wave-uniform => EXEC all 1s
        const int m0 = (t & (mTiles - 1)) << 4;
        const int n0 = (t / mTiles) << 4;
        const int nn = n0 + (lane & 15);
        // A frag: lane<16 -> row m0+lane, K 0..15; lane>=16 -> K 16..31
        const f16* aPtr = A + (m0 + (lane & 15)) * aPitch + ((lane & 16) ? 16 : 0);
        // B frag: lane<16 -> col nn, K 0..15; lane>=16 -> K 16..31
        const f16* bPtr = Wp + nn * 32 + ((lane & 16) ? 16 : 0);
        const int bStep = Np * 32;
        v8f acc = {};
#pragma unroll
        for (int kb = 0; kb < KB; ++kb) {
            v8h alo = *(const v8h*)(aPtr);        // K off+0..7   (ds_load_b128)
            v8h ahi = *(const v8h*)(aPtr + 8);    // K off+8..15
            v8h blo = *(const v8h*)(bPtr);        // global_load_b128 (L2 hot)
            v8h bhi = *(const v8h*)(bPtr + 8);
            if (kb + 1 < KB) __builtin_prefetch(bPtr + bStep, 0, 1);
            v16h av, bvv;
#pragma unroll
            for (int i = 0; i < 8; ++i) {
                av[i] = alo[i];  av[i + 8] = ahi[i];
                bvv[i] = blo[i]; bvv[i + 8] = bhi[i];
            }
            acc = __builtin_amdgcn_wmma_f32_16x16x32_f16(
                false, av, false, bvv, (short)0, acc, false, false);
            aPtr += 32;
            bPtr += bStep;
        }
        const float bb = (nn < Nvalid) ? bias[nn] : 0.0f;
        const int r0 = m0 + ((lane & 16) ? 8 : 0);  // D: reg r -> row r0+r
#pragma unroll
        for (int r = 0; r < 8; ++r) {
            float v = acc[r] + bb;
            if (RELU) v = fmaxf(v, 0.0f);
            if (OUTF32) outF[(r0 + r) * outPitch + nn] = v;
            else        outH[(r0 + r) * outPitch + nn] = (f16)v;
        }
    }
}

// ---------------------------------------------------------------------------
// Fully fused network: 64 rows per workgroup, everything in LDS.
// ---------------------------------------------------------------------------
__global__ void __launch_bounds__(THREADS, 1)
fused_mlp(Params p)
{
    extern __shared__ char smem[];
    f16*   xbuf = (f16*)(smem);
    f16*   hA   = (f16*)(smem + HA_OFF);
    f16*   hB   = (f16*)(smem + HB_OFF);
    f16*   v1b  = (f16*)(smem + V1_OFF);
    f16*   a1b  = (f16*)(smem + A1_OFF);
    f16*   v2b  = (f16*)(smem + V2_OFF);
    f16*   a2b  = (f16*)(smem + A2_OFF);
    float* vF   = (float*)(smem + VF_OFF);
    float* aF   = (float*)(smem + AF_OFF);

    const int tid  = threadIdx.x;
    const int lane = tid & 31;
    const int wave = tid >> 5;
    const int row0 = blockIdx.x * ROWS_PER_WG;

    // ---- Stage x tile: f32 global -> f16 LDS, zero-pad K 937 -> 960(+pad)
    for (int i = tid; i < ROWS_PER_WG * XBUF_PITCH; i += THREADS) {
        int r = i / XBUF_PITCH;
        int c = i - r * XBUF_PITCH;
        float v = (c < 937) ? p.x[(size_t)(row0 + r) * 937 + c] : 0.0f;
        xbuf[r * XBUF_PITCH + c] = (f16)v;
    }
    __syncthreads();

    // ---- Shared trunk
    layer_gemm<30, true, false>(xbuf, XBUF_PITCH, p.w[0], p.b[0], 512, 512, hA, nullptr, H512_PITCH, lane, wave);
    __syncthreads();
    layer_gemm<16, true, false>(hA, H512_PITCH, p.w[1], p.b[1], 512, 512, hB, nullptr, H512_PITCH, lane, wave);
    __syncthreads();
    layer_gemm<16, true, false>(hB, H512_PITCH, p.w[2], p.b[2], 256, 256, hA, nullptr, H256_PITCH, lane, wave);
    __syncthreads();
    layer_gemm<8,  true, false>(hA, H256_PITCH, p.w[3], p.b[3], 256, 256, hB, nullptr, H256_PITCH, lane, wave);
    __syncthreads();
    layer_gemm<8,  true, false>(hB, H256_PITCH, p.w[4], p.b[4], 256, 256, hA, nullptr, H256_PITCH, lane, wave);
    __syncthreads();
    layer_gemm<8,  true, false>(hA, H256_PITCH, p.w[5], p.b[5], 256, 256, hB, nullptr, H256_PITCH, lane, wave);
    __syncthreads();
    layer_gemm<8,  true, false>(hB, H256_PITCH, p.w[6], p.b[6], 256, 256, hA, nullptr, H256_PITCH, lane, wave);
    __syncthreads();                 // trunk h lives in hA (pitch H256)

    // ---- Heads (x is dead; v1b/a1b/v2b/a2b/vF/aF alias xbuf region)
    layer_gemm<8, true, false>(hA, H256_PITCH, p.w[7],  p.b[7],  128, 128, v1b, nullptr, H128_PITCH, lane, wave);
    layer_gemm<8, true, false>(hA, H256_PITCH, p.w[10], p.b[10], 128, 128, a1b, nullptr, H128_PITCH, lane, wave);
    __syncthreads();
    layer_gemm<4, true, false>(v1b, H128_PITCH, p.w[8],  p.b[8],  64, 64, v2b, nullptr, H64_PITCH, lane, wave);
    layer_gemm<4, true, false>(a1b, H128_PITCH, p.w[11], p.b[11], 64, 64, a2b, nullptr, H64_PITCH, lane, wave);
    __syncthreads();
    // Final linears, N padded to 16 (weights zero-padded), f32 out, no ReLU
    layer_gemm<2, false, true>(v2b, H64_PITCH, p.w[9],  p.b[9],  1, 16, nullptr, vF, 16, lane, wave);
    layer_gemm<2, false, true>(a2b, H64_PITCH, p.w[12], p.b[12], 4, 16, nullptr, aF, 16, lane, wave);
    __syncthreads();

    // ---- Dueling combine + suicide/invalid masking (one thread per row)
    if (tid < ROWS_PER_WG) {
        const int  grow  = row0 + tid;
        const float value = vF[tid * 16];
        float adv[4];
        bool  s[4];
        int scnt = 0;
#pragma unroll
        for (int j = 0; j < 4; ++j) {
            adv[j] = aF[tid * 16 + j];
            s[j] = (p.suicides[(size_t)grow * 4 + j] != 0);
            scnt += s[j] ? 1 : 0;
        }
        const bool all_s = (scnt == 4);
        const int  inv   = p.invalid[grow];
        bool  mask[4];
        float denom;
        if (all_s) {
#pragma unroll
            for (int j = 0; j < 4; ++j) mask[j] = (j == inv);
            denom = 3.0f;
        } else {
#pragma unroll
            for (int j = 0; j < 4; ++j) mask[j] = s[j];
            denom = (float)(4 - scnt);
        }
        float sum = 0.0f;
#pragma unroll
        for (int j = 0; j < 4; ++j) if (!mask[j]) sum += adv[j];
        const float mean = sum / denom;
#pragma unroll
        for (int j = 0; j < 4; ++j) {
            float a = mask[j] ? -INFINITY : adv[j];
            p.out[(size_t)grow * 4 + j] = value + (a - mean);
        }
    }
}

// ---------------------------------------------------------------------------
extern "C" void kernel_launch(void* const* d_in, const int* in_sizes, int n_in,
                              void* d_out, int out_size, void* d_ws, size_t ws_size,
                              hipStream_t stream) {
    (void)in_sizes; (void)n_in; (void)out_size; (void)ws_size;

    static const int KT[13] = {937,512,512,256,256,256,256, 256,128,64, 256,128,64};
    static const int NT[13] = {512,512,256,256,256,256,256, 128,64,1,  128,64,4};

    Params p;
    p.x        = (const float*)d_in[0];
    p.suicides = (const unsigned char*)d_in[1];
    p.invalid  = (const int*)d_in[2];
    p.out      = (float*)d_out;

    size_t off = 0;
    for (int i = 0; i < 13; ++i) {
        const int K  = KT[i], N = NT[i];
        const int Kp = ((K + 31) / 32) * 32;
        const int Np = ((N + 15) / 16) * 16;
        f16* wp = (f16*)((char*)d_ws + off);
        size_t bytes = (size_t)Kp * Np * sizeof(f16);
        off += (bytes + 255) & ~(size_t)255;

        p.w[i] = wp;
        p.b[i] = (const float*)d_in[4 + 2 * i];

        const float* W = (const float*)d_in[3 + 2 * i];
        const int total = Kp * Np;
        pack_weights<<<(total + 255) / 256, 256, 0, stream>>>(W, wp, K, N, Kp, Np);
    }

    (void)hipFuncSetAttribute((const void*)fused_mlp,
                              hipFuncAttributeMaxDynamicSharedMemorySize,
                              (int)LDS_TOTAL);
    fused_mlp<<<65536 / ROWS_PER_WG, THREADS, LDS_TOTAL, stream>>>(p);
}
